// RNNEncoder_87969520156816
// MI455X (gfx1250) — matrix operand order repaired and checked by reading, not compiled, not run on hardware
//
#include <hip/hip_runtime.h>
#include <hip/hip_bf16.h>

// ---------------------------------------------------------------------------
// CDNA5 (gfx1250) bidirectional GRU encoder.
//   - All GEMMs: v_wmma_f32_16x16x32_bf16 (bf16 in, fp32 accumulate).
//   - Recurrent scan: persistent workgroup per direction, state in LDS,
//     time-major xp layout, xp latency hidden under the WMMA k-loop via
//     register preloads + global_load_async_to_lds_b128 (ASYNCcnt).
// ---------------------------------------------------------------------------

typedef __attribute__((ext_vector_type(16))) __bf16 bf16x16;
typedef __attribute__((ext_vector_type(8)))  float  f32x8;
typedef __attribute__((ext_vector_type(4)))  unsigned int u32x4;

union FragU {
    bf16x16 v;
    u32x4   q[2];
};

__device__ __forceinline__ unsigned short f2bf(float f) {
    union { float f; unsigned u; } x; x.f = f;
    unsigned r = x.u + 0x7FFFu + ((x.u >> 16) & 1u);   // round-to-nearest-even
    return (unsigned short)(r >> 16);
}

// A fragment, 16x32 (MxK) bf16, ISA §7.12.2 layout:
// lane l: M=l&15, h=l>>4; elems 0-7 -> K=8h..8h+7, elems 8-15 -> K=16+8h..16+8h+7.
__device__ __forceinline__ bf16x16 load_a_frag(const unsigned short* base, int lda,
                                               int M0, int K0, int lane) {
    int m = M0 + (lane & 15);
    int h = (lane >> 4) & 1;
    const unsigned short* p = base + (size_t)m * lda + K0 + 8 * h;
    FragU f;
    f.q[0] = *reinterpret_cast<const u32x4*>(p);
    f.q[1] = *reinterpret_cast<const u32x4*>(p + 16);
    return f.v;
}

// B fragment, 32x16 (KxN) bf16 from row-major weights W(N,K):
// lane l: N=l&15, h=l>>4; elems 0-7 -> K=8h..8h+7, elems 8-15 -> K=16+8h..16+8h+7.
__device__ __forceinline__ bf16x16 load_b_frag(const unsigned short* Wt, int ldb,
                                               int N0, int K0, int lane) {
    int n = N0 + (lane & 15);
    int h = (lane >> 4) & 1;
    const unsigned short* p = Wt + (size_t)n * ldb + K0 + 8 * h;
    FragU f;
    f.q[0] = *reinterpret_cast<const u32x4*>(p);
    f.q[1] = *reinterpret_cast<const u32x4*>(p + 16);
    return f.v;
}

__device__ __forceinline__ f32x8 wmma_bf16(bf16x16 a, bf16x16 b, f32x8 c) {
    return __builtin_amdgcn_wmma_f32_16x16x32_bf16(false, a, false, b,
                                                   (short)0, c, false, false);
}

// ---------------------------------------------------------------------------
__global__ void convert_f32_bf16(const float* __restrict__ in,
                                 unsigned short* __restrict__ out, int n) {
    int i = blockIdx.x * 256 + threadIdx.x;
    if (i < n) out[i] = f2bf(in[i]);
}

__global__ void embed_kernel(const int* __restrict__ x,
                             const float* __restrict__ emb,
                             unsigned short* __restrict__ out) {
    size_t idx = (size_t)blockIdx.x * 256 + threadIdx.x;   // 16384*512 total
    int row = (int)(idx >> 9);
    int col = (int)(idx & 511);
    int tok = x[row];
    out[idx] = f2bf(emb[(size_t)tok * 512 + col]);
}

// ---------------------------------------------------------------------------
// Generic GEMM: C(M,N) fp32 = A(M,K)bf16 @ W(N,K)bf16^T + bias(N)
// block = 256 threads (8 waves, 2x4), wave tile 32x64, block tile 64x256.
// grid = (N/256, M/64).  cswz!=0 stores C time-major: row' = (m&511)*32 + m>>9
// (input rows are b*512+l; output rows become l*32+b).
// ---------------------------------------------------------------------------
__global__ __launch_bounds__(256) void gemm_bf16_nt(
        const unsigned short* __restrict__ A, int lda,
        const unsigned short* __restrict__ W, int ldb,
        const float* __restrict__ bias,
        float* __restrict__ C, int ldc, int K, int cswz) {
    const int lane = threadIdx.x & 31;
    const int wave = threadIdx.x >> 5;
    const int M0 = blockIdx.y * 64 + (wave >> 2) * 32;
    const int N0 = blockIdx.x * 256 + (wave & 3) * 64;

    f32x8 acc[2][4] = {};
    for (int k0 = 0; k0 < K; k0 += 32) {
        if (k0 + 32 < K)   // global_prefetch_b8 for next A k-slab
            __builtin_prefetch(A + (size_t)(M0 + (lane & 15)) * lda + k0 + 32, 0, 1);
        bf16x16 a0 = load_a_frag(A, lda, M0,      k0, lane);
        bf16x16 a1 = load_a_frag(A, lda, M0 + 16, k0, lane);
#pragma unroll
        for (int j = 0; j < 4; ++j) {
            bf16x16 b = load_b_frag(W, ldb, N0 + 16 * j, k0, lane);
            acc[0][j] = wmma_bf16(a0, b, acc[0][j]);
            acc[1][j] = wmma_bf16(a1, b, acc[1][j]);
        }
    }
    const int nl = lane & 15, hh = lane >> 4;
#pragma unroll
    for (int ti = 0; ti < 2; ++ti)
#pragma unroll
        for (int j = 0; j < 4; ++j) {
            int n = N0 + 16 * j + nl;
            float bv = bias ? bias[n] : 0.f;
#pragma unroll
            for (int i = 0; i < 8; ++i) {
                int m = M0 + 16 * ti + 8 * hh + i;
                int mrow = cswz ? (((m & 511) << 5) | (m >> 9)) : m;
                C[(size_t)mrow * ldc + n] = acc[ti][j][i] + bv;
            }
        }
}

// ---------------------------------------------------------------------------
// Persistent GRU scan.  grid = 2 (dir 0=fwd, 1=bwd), block = 512 (16 waves).
// xp is TIME-MAJOR: row = t*32 + b, 1536 cols (bias folded in).
// Per step:
//   Stage A (16 waves): [z|r] = sigmoid(xp[:,0:1024] + s @ U[0:1024]^T)
//     - xp gate values preloaded to registers BEFORE the WMMA k-loop
//     - stage-B xp slice (32x512 f32) copied to LDS with
//       global_load_async_to_lds_b128, waited with s_wait_asynccnt
//     - next timestep's slab prefetched to L2
//     z stays in registers; r -> s*r stored to LDS bf16 by waves 8..15.
//   Stage B (waves 0..7): s_hat = relu(xs_lds + sr @ Us^T); state update is
//     register-local (stage-B tiling == stage-A z tiling).
// ---------------------------------------------------------------------------
#define SF_LD 516   // fp32 row stride (pad to dodge bank conflicts)
#define SB_LD 520   // bf16 row stride (1040 B, 16B aligned)
#define SMEM_SCAN (32*SF_LD*4 + 2*32*SB_LD*2 + 32*SF_LD*4)   // ~194 KB < 320 KB

__global__ __launch_bounds__(512) void gru_scan(
        const float* __restrict__ xp_f, const float* __restrict__ xp_b,
        const unsigned short* __restrict__ U_f, const unsigned short* __restrict__ U_b,
        const float* __restrict__ mask,
        unsigned short* __restrict__ out_bf, float* __restrict__ out_f32) {
    const int  dir     = blockIdx.x;
    const bool reverse = (dir == 1);
    const float*          xp = reverse ? xp_b : xp_f;
    const unsigned short* U  = reverse ? U_b  : U_f;
    const int  dofs = dir * 512;

    extern __shared__ char smem[];
    float*          s_st = (float*)smem;                                   // [32][SF_LD]
    unsigned short* s_b  = (unsigned short*)(smem + 32*SF_LD*4);           // [32][SB_LD]
    unsigned short* sr_b = (unsigned short*)(smem + 32*SF_LD*4 + 32*SB_LD*2);
    float*          xs   = (float*)(smem + 32*SF_LD*4 + 2*32*SB_LD*2);     // [32][SF_LD]

    const int lane = threadIdx.x & 31;
    const int wave = threadIdx.x >> 5;
    const int nl = lane & 15, hh = lane >> 4;
    const int N0w = wave * 64;          // stage-A column block in [0,1024)
    const int N0s = (wave & 7) * 64;    // stage-B column block in [0,512)
    const int sm  = threadIdx.x >> 4;        // async-copy: batch row 0..31
    const int sc  = (threadIdx.x & 15) * 32; // async-copy: col chunk in [0,512)

    for (int idx = threadIdx.x; idx < 32 * 512; idx += 512) {
        int m = idx >> 9, c = idx & 511;
        s_st[m * SF_LD + c] = 0.f;
        s_b [m * SB_LD + c] = 0;
    }

    f32x8 zreg[2][4];   // z gate, meaningful on waves 0..7 only

    for (int step = 0; step < 512; ++step) {
        const int t = reverse ? (511 - step) : step;
        __syncthreads();

        const float* slab = xp + (size_t)t * 32 * 1536;   // contiguous 192KB

        // ---- async copy of stage-B xp slice (cols 1024..1535) into LDS ----
        {
            const float* gsrc = slab + (size_t)sm * 1536 + 1024 + sc;  // 32 floats
            unsigned ldsa = (unsigned)(uintptr_t)(xs + sm * SF_LD + sc);
#pragma unroll
            for (int q = 0; q < 8; ++q) {
                asm volatile("global_load_async_to_lds_b128 %0, %1, off"
                             :: "v"(ldsa + q * 16), "v"((const void*)(gsrc + q * 4))
                             : "memory");
            }
        }

        // ---- preload stage-A xp gate values into registers ----
        float xpvA[2][4][8];
#pragma unroll
        for (int ti = 0; ti < 2; ++ti)
#pragma unroll
            for (int j = 0; j < 4; ++j) {
                const int n = N0w + 16 * j + nl;
#pragma unroll
                for (int i = 0; i < 8; ++i) {
                    const int m = 16 * ti + 8 * hh + i;
                    xpvA[ti][j][i] = slab[(size_t)m * 1536 + n];
                }
            }

        // ---- prefetch next timestep's slab into L2 ----
        if (step + 1 < 512) {
            const int tn = reverse ? (t - 1) : (t + 1);
            const float* nslab = xp + (size_t)tn * 32 * 1536;
#pragma unroll
            for (int q = 0; q < 3; ++q)
                __builtin_prefetch(nslab + ((size_t)threadIdx.x * 3 + q) * 32, 0, 1);
        }

        // ---- Stage A GEMM: acc = s @ U[0:1024]^T ----
        f32x8 acc[2][4] = {};
        for (int k0 = 0; k0 < 512; k0 += 32) {
            bf16x16 a0 = load_a_frag(s_b, SB_LD, 0,  k0, lane);
            bf16x16 a1 = load_a_frag(s_b, SB_LD, 16, k0, lane);
#pragma unroll
            for (int j = 0; j < 4; ++j) {
                bf16x16 b = load_b_frag(U, 512, N0w + 16 * j, k0, lane);
                acc[0][j] = wmma_bf16(a0, b, acc[0][j]);
                acc[1][j] = wmma_bf16(a1, b, acc[1][j]);
            }
        }
        const bool is_z = (wave < 8);
#pragma unroll
        for (int ti = 0; ti < 2; ++ti)
#pragma unroll
            for (int j = 0; j < 4; ++j) {
                const int n = N0w + 16 * j + nl;
#pragma unroll
                for (int i = 0; i < 8; ++i) {
                    const int m = 16 * ti + 8 * hh + i;
                    float v = acc[ti][j][i] + xpvA[ti][j][i];
                    float g = 1.f / (1.f + __expf(-v));
                    if (is_z) {
                        zreg[ti][j][i] = g;                     // z gate
                    } else {
                        int c = n - 512;                         // r gate
                        sr_b[m * SB_LD + c] = f2bf(g * s_st[m * SF_LD + c]);
                    }
                }
            }
        asm volatile("s_wait_asynccnt 0" ::: "memory");   // xs slab resident
        __syncthreads();

        // ---- Stage B: s_hat = relu(xs + sr @ Us^T); state update ----
        if (wave < 8) {
            f32x8 acc2[2][4] = {};
            const unsigned short* Us = U + 1024 * 512;
            for (int k0 = 0; k0 < 512; k0 += 32) {
                bf16x16 a0 = load_a_frag(sr_b, SB_LD, 0,  k0, lane);
                bf16x16 a1 = load_a_frag(sr_b, SB_LD, 16, k0, lane);
#pragma unroll
                for (int j = 0; j < 4; ++j) {
                    bf16x16 b = load_b_frag(Us, 512, N0s + 16 * j, k0, lane);
                    acc2[0][j] = wmma_bf16(a0, b, acc2[0][j]);
                    acc2[1][j] = wmma_bf16(a1, b, acc2[1][j]);
                }
            }
#pragma unroll
            for (int ti = 0; ti < 2; ++ti)
#pragma unroll
                for (int j = 0; j < 4; ++j) {
                    const int n = N0s + 16 * j + nl;
#pragma unroll
                    for (int i = 0; i < 8; ++i) {
                        const int m = 16 * ti + 8 * hh + i;
                        float v  = acc2[ti][j][i] + xs[m * SF_LD + n];
                        float sh = v > 0.f ? v : 0.f;
                        float z  = zreg[ti][j][i];
                        float so = s_st[m * SF_LD + n];
                        float sn = (1.f - z) * so + z * sh;
                        float mk = mask[m * 512 + t];
                        float o  = so * (1.f - mk) + sn * mk;
                        s_st[m * SF_LD + n] = o;
                        unsigned short ob = f2bf(o);
                        s_b[m * SB_LD + n] = ob;
                        size_t oidx = ((size_t)m * 512 + t) * 1024 + dofs + n;
                        out_bf[oidx] = ob;
                        if (out_f32) out_f32[oidx] = o;
                    }
                }
        }
    }
}

// ---------------------------------------------------------------------------
// Host orchestration.
// Inputs: 0=x 1=mask 2=emb_W 3=U_attr 4=W0f 5=U0f 6=b0f 7=W0b 8=U0b 9=b0b
//         10=W1f 11=U1f 12=b1f 13=W1b 14=U1b 15=b1b
// d_out = [h (16384x1024 f32) | enc_keys (16384x512 f32)]
// ---------------------------------------------------------------------------
extern "C" void kernel_launch(void* const* d_in, const int* in_sizes, int n_in,
                              void* d_out, int out_size, void* d_ws, size_t ws_size,
                              hipStream_t stream) {
    (void)in_sizes; (void)n_in; (void)out_size; (void)ws_size;

    const int*   x     = (const int*)  d_in[0];
    const float* mask  = (const float*)d_in[1];
    const float* embW  = (const float*)d_in[2];
    const float* Uattr = (const float*)d_in[3];
    const float* W0f = (const float*)d_in[4],  *U0f = (const float*)d_in[5],  *b0f = (const float*)d_in[6];
    const float* W0b = (const float*)d_in[7],  *U0b = (const float*)d_in[8],  *b0b = (const float*)d_in[9];
    const float* W1f = (const float*)d_in[10], *U1f = (const float*)d_in[11], *b1f = (const float*)d_in[12];
    const float* W1b = (const float*)d_in[13], *U1b = (const float*)d_in[14], *b1b = (const float*)d_in[15];

    // -------- workspace layout --------
    unsigned short* wsu   = (unsigned short*)d_ws;
    unsigned short* bW0f  = wsu;                 // 1536x512
    unsigned short* bU0f  = wsu + 786432;
    unsigned short* bW0b  = wsu + 1572864;
    unsigned short* bU0b  = wsu + 2359296;
    unsigned short* bW1f  = wsu + 3145728;       // 1536x1024
    unsigned short* bU1f  = wsu + 4718592;
    unsigned short* bW1b  = wsu + 5505024;       // 1536x1024
    unsigned short* bU1b  = wsu + 7077888;
    unsigned short* bUat  = wsu + 7864320;       // 512x1024
    char* wsb = (char*)d_ws;
    unsigned short* bufA = (unsigned short*)(wsb + 16777216);    // 16384x1024 bf16
    unsigned short* bufB = (unsigned short*)(wsb + 50331648);    // 16384x1024 bf16
    float* xpf = (float*)(wsb + 83886080);                       // [512][32][1536] f32
    float* xpb = (float*)(wsb + 184549376);                      // [512][32][1536] f32

    auto conv = [&](const float* s, unsigned short* dptr, int n) {
        convert_f32_bf16<<<(n + 255) / 256, 256, 0, stream>>>(s, dptr, n);
    };
    conv(W0f, bW0f, 786432);  conv(U0f, bU0f, 786432);
    conv(W0b, bW0b, 786432);  conv(U0b, bU0b, 786432);
    conv(W1f, bW1f, 1572864); conv(U1f, bU1f, 786432);
    conv(W1b, bW1b, 1572864); conv(U1b, bU1b, 786432);
    conv(Uattr, bUat, 524288);

    // embedding gather -> bufA (bf16)
    embed_kernel<<<32768, 256, 0, stream>>>(x, embW, bufA);

    dim3 gXP(1536 / 256, 16384 / 64);
    // layer 0 input projections (bias folded in, time-major output)
    gemm_bf16_nt<<<gXP, 256, 0, stream>>>(bufA, 512, bW0f, 512, b0f, xpf, 1536, 512, 1);
    gemm_bf16_nt<<<gXP, 256, 0, stream>>>(bufA, 512, bW0b, 512, b0b, xpb, 1536, 512, 1);
    // layer 0 scan -> bufB (bf16 concat fwd|bwd)
    gru_scan<<<2, 512, SMEM_SCAN, stream>>>(xpf, xpb, bU0f, bU0b, mask, bufB, nullptr);

    // layer 1 input projections
    gemm_bf16_nt<<<gXP, 256, 0, stream>>>(bufB, 1024, bW1f, 1024, b1f, xpf, 1536, 1024, 1);
    gemm_bf16_nt<<<gXP, 256, 0, stream>>>(bufB, 1024, bW1b, 1024, b1b, xpb, 1536, 1024, 1);
    // layer 1 scan -> h (fp32, d_out) + bufA (bf16 for attention GEMM)
    float* hout = (float*)d_out;
    gru_scan<<<2, 512, SMEM_SCAN, stream>>>(xpf, xpb, bU1f, bU1b, mask, bufA, hout);

    // enc_keys = h @ U_attr^T
    gemm_bf16_nt<<<dim3(512 / 256, 16384 / 64), 256, 0, stream>>>(
        bufA, 1024, bUat, 1024, nullptr, hout + (size_t)16384 * 1024, 512, 1024, 0);
}
